// MSDeformAttnTransformerEncoderOnly_9680856285300
// MI455X (gfx1250) — matrix operand order repaired and checked by reading, not compiled, not run on hardware
//
#include <hip/hip_runtime.h>
#include <hip/hip_bf16.h>

typedef unsigned short u16;
typedef __attribute__((ext_vector_type(16))) __bf16 v16bf;
typedef __attribute__((ext_vector_type(8)))  float  v8f;

#define M_TOT   10752     // B * S
#define S_TOT   5376
#define D_MODEL 256

__device__ __forceinline__ u16 f32_to_bf16(float f) {
    unsigned int u = __float_as_uint(f);
    u += 0x7FFFu + ((u >> 16) & 1u);     // round-to-nearest-even
    return (u16)(u >> 16);
}

union Frag { uint4 u[2]; v16bf v; };

// CDNA5 async global->LDS copy (per-lane, tracked by ASYNCcnt).
// Low 32 bits of a generic LDS pointer are the LDS byte offset (ISA 10.2).
__device__ __forceinline__ void async_copy_b128(void* lds_dst, const void* gaddr) {
    asm volatile("global_load_async_to_lds_b128 %0, %1, off"
                 :: "v"((unsigned)(size_t)lds_dst), "v"(gaddr)
                 : "memory");
}
__device__ __forceinline__ void wait_async0() {
    asm volatile("s_wait_asynccnt 0x0" ::: "memory");
}

// ---------------------------------------------------------------------------
// WMMA bf16 GEMM: out[M,N] = A[M,K] @ Wt[N,K]^T + bias
// MODE 0: f32 store; MODE 1: f32 accumulate (residual); MODE 2: bf16 relu store
// Block: 256 thr = 8 waves (4 M x 2 N); block tile 128x128; wave tile 32x64 =
// 2x4 v_wmma_f32_16x16x32_bf16. Double-buffered LDS staged with
// global_load_async_to_lds_b128 + s_wait_asynccnt. Requires M % 128 == 0.
// ---------------------------------------------------------------------------
template<int MODE>
__global__ __launch_bounds__(256)
void gemm_bf16_wmma(const u16* __restrict__ A, const u16* __restrict__ Wt,
                    const float* __restrict__ bias, void* __restrict__ outv,
                    int M, int N, int K)
{
    __shared__ u16 As[2][128 * 32];   // [buf][m][k] 8 KB each
    __shared__ u16 Bs[2][128 * 32];   // [buf][n][k] 8 KB each

    const int tid  = threadIdx.x;
    const int lane = tid & 31;
    const int wave = tid >> 5;
    const int wm   = wave & 3;          // wave M index 0..3
    const int wn   = wave >> 2;         // wave N index 0..1
    const int m0   = blockIdx.y * 128;
    const int n0   = blockIdx.x * 128;

    v8f acc[2][4] = {};

    // staging: each thread moves 16 contiguous bf16 (2x b128) for A and B
    const int srow  = tid >> 1;         // 0..127
    const int shalf = (tid & 1) * 16;   // 0 | 16 (elements)

    const int ga_row = m0 + srow;                         // M % 128 == 0 -> in range
    int gb_row = n0 + srow;
    if (gb_row >= N) gb_row = N - 1;   // clamp: junk lands only in cols >= N (discarded)

    const u16* aptr = A  + (size_t)ga_row * K + shalf;
    const u16* bptr = Wt + (size_t)gb_row * K + shalf;
    u16* as_dst = &As[0][srow * 32 + shalf];
    u16* bs_dst = &Bs[0][srow * 32 + shalf];
    const size_t bufstride = 128 * 32;   // elements between buf 0 and buf 1

    auto stage = [&](int buf, int k0) {
        async_copy_b128(as_dst + (size_t)buf * bufstride,     aptr + k0);
        async_copy_b128(as_dst + (size_t)buf * bufstride + 8, aptr + k0 + 8);
        async_copy_b128(bs_dst + (size_t)buf * bufstride,     bptr + k0);
        async_copy_b128(bs_dst + (size_t)buf * bufstride + 8, bptr + k0 + 8);
    };

    stage(0, 0);
    wait_async0();
    __syncthreads();

    const int ntile = K >> 5;
    for (int t = 0; t < ntile; ++t) {
        const int buf = t & 1;
        if (t + 1 < ntile) stage(buf ^ 1, (t + 1) * 32);   // overlap DMA with WMMA

        // --- fragments per ISA 7.12.2 16-bit layouts ---
        Frag afr[2], bfr[4];
        const int akb = (lane >> 4) * 8;    // A: lane<16 -> K 0-7/16-23 ; lane>=16 -> K 8-15/24-31
        const int am  = lane & 15;
        const u16* asb = &As[buf][0];
        #pragma unroll
        for (int mi = 0; mi < 2; ++mi) {
            int r = wm * 32 + mi * 16 + am;
            afr[mi].u[0] = *reinterpret_cast<const uint4*>(asb + r * 32 + akb);
            afr[mi].u[1] = *reinterpret_cast<const uint4*>(asb + r * 32 + akb + 16);
        }
        const int bkb = (lane >> 4) * 8;    // B: lane = N, lane<16 -> K 0-7/16-23 ; lane>=16 -> K 8-15/24-31
        const u16* bsb = &Bs[buf][0];
        #pragma unroll
        for (int ni = 0; ni < 4; ++ni) {
            int c = wn * 64 + ni * 16 + (lane & 15);
            bfr[ni].u[0] = *reinterpret_cast<const uint4*>(bsb + c * 32 + bkb);
            bfr[ni].u[1] = *reinterpret_cast<const uint4*>(bsb + c * 32 + bkb + 16);
        }
        #pragma unroll
        for (int mi = 0; mi < 2; ++mi)
            #pragma unroll
            for (int ni = 0; ni < 4; ++ni)
                acc[mi][ni] = __builtin_amdgcn_wmma_f32_16x16x32_bf16(
                    false, afr[mi].v, false, bfr[ni].v, (short)0, acc[mi][ni], false, false);

        wait_async0();       // next tile landed in LDS (all waves own their issues)
        __syncthreads();     // all waves done reading buf before it is re-staged
    }

    // --- epilogue: C/D layout lane<16: M=i, lane>=16: M=i+8; N=lane&15 ---
    const int rb = (lane >> 4) * 8;
    const int cn = lane & 15;
    #pragma unroll
    for (int mi = 0; mi < 2; ++mi)
        #pragma unroll
        for (int ni = 0; ni < 4; ++ni) {
            int col = n0 + wn * 64 + ni * 16 + cn;
            if (col >= N) continue;
            float bvv = bias ? bias[col] : 0.0f;
            #pragma unroll
            for (int i = 0; i < 8; ++i) {
                int row = m0 + wm * 32 + mi * 16 + rb + i;
                float v = acc[mi][ni][i] + bvv;
                size_t o = (size_t)row * N + col;
                if (MODE == 0)      ((float*)outv)[o] = v;
                else if (MODE == 1) ((float*)outv)[o] += v;
                else                ((u16*)outv)[o] = f32_to_bf16(v > 0.f ? v : 0.f);
            }
        }
}

// ---------------------------------------------------------------------------
// LayerNorm -> h (bf16), optionally q = h + pos(level) (bf16)
// ---------------------------------------------------------------------------
__global__ __launch_bounds__(256)
void ln_kernel(const float* __restrict__ x, const float* __restrict__ g,
               const float* __restrict__ beta, const float* __restrict__ level_embed,
               u16* __restrict__ h_out, u16* __restrict__ q_out)
{
    __shared__ float red[256];
    const int row = blockIdx.x;
    const int t = threadIdx.x;
    float xv = x[(size_t)row * D_MODEL + t];
    red[t] = xv; __syncthreads();
    #pragma unroll
    for (int s2 = 128; s2 > 0; s2 >>= 1) { if (t < s2) red[t] += red[t + s2]; __syncthreads(); }
    float mu = red[0] * (1.0f / D_MODEL);
    __syncthreads();
    float dv = xv - mu;
    red[t] = dv * dv; __syncthreads();
    #pragma unroll
    for (int s2 = 128; s2 > 0; s2 >>= 1) { if (t < s2) red[t] += red[t + s2]; __syncthreads(); }
    float var = red[0] * (1.0f / D_MODEL);
    float y = dv * rsqrtf(var + 1e-6f) * g[t] + beta[t];
    h_out[(size_t)row * D_MODEL + t] = f32_to_bf16(y);
    if (q_out) {
        int s = row % S_TOT;
        int l = (s < 4096) ? 0 : ((s < 5120) ? 1 : 2);
        q_out[(size_t)row * D_MODEL + t] = f32_to_bf16(y + level_embed[l * D_MODEL + t]);
    }
}

// ---------------------------------------------------------------------------
// Deformable sampling: softmax(12 pts/head) + bilinear gather of v, bf16 out
// One block per (b,s): thread t -> head = t/32, dim = t%32
// ---------------------------------------------------------------------------
__global__ __launch_bounds__(256)
void deform_sample_kernel(const float* __restrict__ off, const float* __restrict__ awl,
                          const float* __restrict__ v, const float* __restrict__ refp,
                          u16* __restrict__ out)
{
    __shared__ float slog[96], sw[96], spx[96], spy[96];
    constexpr int LW[3] = {64, 32, 16};
    constexpr int LH[3] = {64, 32, 16};
    constexpr int LST[3] = {0, 4096, 5120};

    const int row = blockIdx.x;          // b*S + s
    const int s = row % S_TOT;
    const int b = row / S_TOT;
    const int t = threadIdx.x;

    if (t < 96) {
        slog[t] = awl[(size_t)row * 96 + t];
        float ox = off[(size_t)row * 192 + 2 * t];
        float oy = off[(size_t)row * 192 + 2 * t + 1];
        int l = (t % 12) >> 2;
        float Wf = (float)LW[l], Hf = (float)LH[l];
        float lx = refp[2 * s]     + ox / Wf;   // wh = (W, H)
        float ly = refp[2 * s + 1] + oy / Hf;
        spx[t] = lx * Wf - 0.5f;
        spy[t] = ly * Hf - 0.5f;
    }
    __syncthreads();
    if (t < 8) {
        float mx = -1e30f;
        for (int r = 0; r < 12; ++r) mx = fmaxf(mx, slog[t * 12 + r]);
        float e[12], sum = 0.f;
        for (int r = 0; r < 12; ++r) { e[r] = __expf(slog[t * 12 + r] - mx); sum += e[r]; }
        float inv = 1.0f / sum;
        for (int r = 0; r < 12; ++r) sw[t * 12 + r] = e[r] * inv;
    }
    __syncthreads();

    const int h = t >> 5, d = t & 31;
    float acc = 0.f;
    #pragma unroll
    for (int r = 0; r < 12; ++r) {
        int p = h * 12 + r;
        int l = r >> 2;
        int Wl = LW[l], Hl = LH[l];
        float px = spx[p], py = spy[p];
        float x0f = floorf(px), y0f = floorf(py);
        int x0 = (int)x0f, y0 = (int)y0f;
        float wx1 = px - x0f, wy1 = py - y0f;
        float wx0 = 1.f - wx1, wy0 = 1.f - wy1;
        const float* vb = v + ((size_t)b * S_TOT + LST[l]) * D_MODEL + h * 32 + d;
        float sval = 0.f;
        #pragma unroll
        for (int dy = 0; dy < 2; ++dy)
            #pragma unroll
            for (int dx = 0; dx < 2; ++dx) {
                int xi = x0 + dx, yi = y0 + dy;
                if (xi >= 0 && xi < Wl && yi >= 0 && yi < Hl) {
                    float w = (dx ? wx1 : wx0) * (dy ? wy1 : wy0);
                    sval += w * vb[(size_t)(yi * Wl + xi) * D_MODEL];
                }
            }
        acc += sw[p] * sval;
    }
    out[(size_t)row * D_MODEL + h * 32 + d] = f32_to_bf16(acc);
}

// ---------------------------------------------------------------------------
__global__ void ref_points_kernel(float* __restrict__ refp) {
    int s = blockIdx.x * 256 + threadIdx.x;
    if (s >= S_TOT) return;
    constexpr int LW[3] = {64, 32, 16};
    constexpr int LH[3] = {64, 32, 16};
    constexpr int LST[3] = {0, 4096, 5120};
    int l = (s < 4096) ? 0 : ((s < 5120) ? 1 : 2);
    int i = s - LST[l];
    int yi = i / LW[l], xi = i % LW[l];
    refp[2 * s]     = (xi + 0.5f) / (float)LW[l];
    refp[2 * s + 1] = (yi + 0.5f) / (float)LH[l];
}

__global__ void wtrans_kernel(const float* __restrict__ w, u16* __restrict__ wt, int K, int N) {
    int idx = blockIdx.x * 256 + threadIdx.x;
    if (idx >= K * N) return;
    int k = idx / N, n = idx % N;
    wt[(size_t)n * K + k] = f32_to_bf16(w[idx]);
}

__global__ void copy_f32_kernel(const float* __restrict__ in, float* __restrict__ out, int n) {
    int i = blockIdx.x * 256 + threadIdx.x;
    if (i < n) out[i] = in[i];
}

// ---------------------------------------------------------------------------
extern "C" void kernel_launch(void* const* d_in, const int* in_sizes, int n_in,
                              void* d_out, int out_size, void* d_ws, size_t ws_size,
                              hipStream_t stream) {
    (void)in_sizes; (void)n_in; (void)out_size; (void)ws_size;
    const float* src     = (const float*)d_in[0];
    const float* lvl_emb = (const float*)d_in[1];
    const float* ln1_g   = (const float*)d_in[2];
    const float* ln1_b   = (const float*)d_in[3];
    const float* off_w   = (const float*)d_in[4];
    const float* off_b   = (const float*)d_in[5];
    const float* aw_w    = (const float*)d_in[6];
    const float* aw_b    = (const float*)d_in[7];
    const float* vp_w    = (const float*)d_in[8];
    const float* vp_b    = (const float*)d_in[9];
    const float* op_w    = (const float*)d_in[10];
    const float* op_b    = (const float*)d_in[11];
    const float* ln2_g   = (const float*)d_in[12];
    const float* ln2_b   = (const float*)d_in[13];
    const float* fc1_w   = (const float*)d_in[14];
    const float* fc1_b   = (const float*)d_in[15];
    const float* fc2_w   = (const float*)d_in[16];
    const float* fc2_b   = (const float*)d_in[17];

    float* x = (float*)d_out;            // residual stream lives in the output
    const int M = M_TOT, D = D_MODEL;

    size_t woff = 0;
    auto carve = [&](size_t bytes) -> void* {
        void* p = (char*)d_ws + woff;
        woff += (bytes + 255) & ~(size_t)255;
        return p;
    };
    float* refp   = (float*)carve((size_t)S_TOT * 2 * 4);
    u16*   h_bf   = (u16*)  carve((size_t)M * D * 2);
    u16*   q_bf   = (u16*)  carve((size_t)M * D * 2);
    u16*   h2_bf  = (u16*)  carve((size_t)M * D * 2);
    float* v_f    = (float*)carve((size_t)M * D * 4);
    float* off_f  = (float*)carve((size_t)M * 192 * 4);
    float* aw_f   = (float*)carve((size_t)M * 96 * 4);
    u16*   at_bf  = (u16*)  carve((size_t)M * D * 2);
    u16*   ffn_bf = (u16*)  carve((size_t)M * 1024 * 2);
    u16*   w_vp   = (u16*)  carve((size_t)256 * 256 * 2);
    u16*   w_off  = (u16*)  carve((size_t)256 * 192 * 2);
    u16*   w_aw   = (u16*)  carve((size_t)256 * 96 * 2);
    u16*   w_op   = (u16*)  carve((size_t)256 * 256 * 2);
    u16*   w_fc1  = (u16*)  carve((size_t)256 * 1024 * 2);
    u16*   w_fc2  = (u16*)  carve((size_t)1024 * 256 * 2);

    ref_points_kernel<<<(S_TOT + 255) / 256, 256, 0, stream>>>(refp);
    copy_f32_kernel<<<(M * D + 255) / 256, 256, 0, stream>>>(src, x, M * D);

    const dim3 blk(256);
    const int MT = M / 128;   // 84
    for (int i = 0; i < 3; ++i) {
        ln_kernel<<<M, blk, 0, stream>>>(x, ln1_g + i * D, ln1_b + i * D, lvl_emb, h_bf, q_bf);

        wtrans_kernel<<<(256 * 256 + 255) / 256, blk, 0, stream>>>(vp_w  + (size_t)i * 256 * 256,  w_vp,  256, 256);
        wtrans_kernel<<<(256 * 192 + 255) / 256, blk, 0, stream>>>(off_w + (size_t)i * 256 * 192,  w_off, 256, 192);
        wtrans_kernel<<<(256 *  96 + 255) / 256, blk, 0, stream>>>(aw_w  + (size_t)i * 256 * 96,   w_aw,  256, 96);
        wtrans_kernel<<<(256 * 256 + 255) / 256, blk, 0, stream>>>(op_w  + (size_t)i * 256 * 256,  w_op,  256, 256);
        wtrans_kernel<<<(256 * 1024 + 255) / 256, blk, 0, stream>>>(fc1_w + (size_t)i * 256 * 1024, w_fc1, 256, 1024);
        wtrans_kernel<<<(1024 * 256 + 255) / 256, blk, 0, stream>>>(fc2_w + (size_t)i * 1024 * 256, w_fc2, 1024, 256);

        gemm_bf16_wmma<0><<<dim3(2, MT), blk, 0, stream>>>(q_bf, w_off, off_b + i * 192, off_f, M, 192, 256);
        gemm_bf16_wmma<0><<<dim3(1, MT), blk, 0, stream>>>(q_bf, w_aw,  aw_b  + i * 96,  aw_f,  M, 96,  256);
        gemm_bf16_wmma<0><<<dim3(2, MT), blk, 0, stream>>>(h_bf, w_vp,  vp_b  + i * D,   v_f,   M, 256, 256);

        deform_sample_kernel<<<M, blk, 0, stream>>>(off_f, aw_f, v_f, refp, at_bf);

        gemm_bf16_wmma<1><<<dim3(2, MT), blk, 0, stream>>>(at_bf, w_op, op_b + i * D, x, M, 256, 256);

        ln_kernel<<<M, blk, 0, stream>>>(x, ln2_g + i * D, ln2_b + i * D, nullptr, h2_bf, nullptr);

        gemm_bf16_wmma<2><<<dim3(8, MT), blk, 0, stream>>>(h2_bf, w_fc1, fc1_b + i * 1024, ffn_bf, M, 1024, 256);
        gemm_bf16_wmma<1><<<dim3(2, MT), blk, 0, stream>>>(ffn_bf, w_fc2, fc2_b + i * D,   x,      M, 256, 1024);
    }
}
